// SimpleTransformer_70643622084859
// MI455X (gfx1250) — compile-verified, matrix operand order checked by reading, not run on hardware
//
#include <hip/hip_runtime.h>

typedef __attribute__((ext_vector_type(16))) _Float16 v16h;
typedef __attribute__((ext_vector_type(8)))  float    v8f;

union H2 { _Float16 h[2]; unsigned int u; };

__device__ __forceinline__ float fast_sigmoid(float z) {
    // 1/(1+e^-z) with single v_rcp_f32 instead of IEEE div chain
    return __builtin_amdgcn_rcpf(1.0f + __expf(-z));
}

// ---------------------------------------------------------------------------
// SetPhi: t in [0,1] -> MLP(1->64->64->64->32), one thread per t value.
// ---------------------------------------------------------------------------
__global__ __launch_bounds__(64) void setphi_kernel(
    const float* __restrict__ PW1, const float* __restrict__ Pb1,
    const float* __restrict__ PW2, const float* __restrict__ Pb2,
    const float* __restrict__ PW3, const float* __restrict__ Pb3,
    const float* __restrict__ PW4, const float* __restrict__ Pb4,
    float* __restrict__ SetPhi)
{
    int i = blockIdx.x * blockDim.x + threadIdx.x;   // 0..1023
    float t = (float)i * (1.0f / 1023.0f);
    float h1[64], h2[64];
    #pragma unroll 4
    for (int c = 0; c < 64; ++c) h1[c] = fmaxf(PW1[c] * t + Pb1[c], 0.0f);
    for (int c = 0; c < 64; ++c) {
        float a = Pb2[c]; const float* w = PW2 + c * 64;
        #pragma unroll 8
        for (int d = 0; d < 64; ++d) a += w[d] * h1[d];
        h2[c] = fmaxf(a, 0.0f);
    }
    for (int c = 0; c < 64; ++c) {
        float a = Pb3[c]; const float* w = PW3 + c * 64;
        #pragma unroll 8
        for (int d = 0; d < 64; ++d) a += w[d] * h2[d];
        h1[c] = fmaxf(a, 0.0f);
    }
    for (int k = 0; k < 32; ++k) {
        float a = Pb4[k]; const float* w = PW4 + k * 64;
        #pragma unroll 8
        for (int d = 0; d < 64; ++d) a += w[d] * h1[d];
        SetPhi[i * 32 + k] = a;
    }
}

// ---------------------------------------------------------------------------
// V MLP: per (b,i) row, 16->16 relu ->16 relu ->8 sigmoid.
// ---------------------------------------------------------------------------
__global__ __launch_bounds__(256) void vproj_kernel(
    const float* __restrict__ x,
    const float* __restrict__ VW1, const float* __restrict__ Vb1,
    const float* __restrict__ VW2, const float* __restrict__ Vb2,
    const float* __restrict__ VW3, const float* __restrict__ Vb3,
    float* __restrict__ vbuf)
{
    int idx = blockIdx.x * blockDim.x + threadIdx.x;  // 0..8191 = b*1024+i
    const float* xr = x + idx * 16;
    float xv[16], a1[16], a2[16];
    #pragma unroll
    for (int d = 0; d < 16; ++d) xv[d] = xr[d];
    #pragma unroll
    for (int c = 0; c < 16; ++c) {
        float a = Vb1[c]; const float* w = VW1 + c * 16;
        #pragma unroll
        for (int d = 0; d < 16; ++d) a += w[d] * xv[d];
        a1[c] = fmaxf(a, 0.0f);
    }
    #pragma unroll
    for (int c = 0; c < 16; ++c) {
        float a = Vb2[c]; const float* w = VW2 + c * 16;
        #pragma unroll
        for (int d = 0; d < 16; ++d) a += w[d] * a1[d];
        a2[c] = fmaxf(a, 0.0f);
    }
    #pragma unroll
    for (int c = 0; c < 8; ++c) {
        float a = Vb3[c]; const float* w = VW3 + c * 16;
        #pragma unroll
        for (int d = 0; d < 16; ++d) a += w[d] * a2[d];
        vbuf[idx * 8 + c] = fast_sigmoid(a);
    }
}

// ---------------------------------------------------------------------------
// Project x by W (64x16) -> f16, swizzled directly into CDNA5 WMMA 16-bit
// 16x32 fragment layout (ISA 7.12.2): lane L, VGPR v (dword):
//   row  = L & 15
//   kbase = chunk*32 + (v&3)*2 + (v>>2)*16 + (L>=16 ? 8 : 0)
// dword = pack(f16(val[kbase]), f16(val[kbase+1])). 16 dwords per lane
// (two K=32 chunks). One wave per 16-row tile.
// ---------------------------------------------------------------------------
__global__ __launch_bounds__(32) void pack_qk(
    const float* __restrict__ x, const float* __restrict__ W,
    unsigned int* __restrict__ packed)
{
    int tile = blockIdx.x & 63;
    int b    = blockIdx.x >> 6;
    int lane = threadIdx.x;
    int m    = lane & 15;
    const float* xr = x + ((b * 1024) + tile * 16 + m) * 16;
    float xv[16];
    #pragma unroll
    for (int d = 0; d < 16; ++d) xv[d] = xr[d];
    unsigned int* outp = packed + (((b * 64) + tile) * 32 + lane) * 16;
    int hi = (lane >> 4) * 8;
    #pragma unroll
    for (int chunk = 0; chunk < 2; ++chunk) {
        #pragma unroll
        for (int v = 0; v < 8; ++v) {
            int kbase = chunk * 32 + ((v & 3) * 2) + ((v >> 2) * 16) + hi;
            const float* w0 = W + kbase * 16;
            float s0 = 0.0f, s1 = 0.0f;
            #pragma unroll
            for (int d = 0; d < 16; ++d) { s0 += xv[d] * w0[d]; s1 += xv[d] * w0[16 + d]; }
            H2 cv; cv.h[0] = (_Float16)s0; cv.h[1] = (_Float16)s1;
            outp[chunk * 8 + v] = cv.u;
        }
    }
}

// ---------------------------------------------------------------------------
// Main fused kernel: per (b, i-tile of 16). 8 waves, each wave covers 8
// j-tiles. Score tile via 2 chained v_wmma_f32_16x16x32_f16 (K=64), fused
// fast-sigmoid(a/8 + offset) with zeroed diagonal, then per-lane accumulation
// of set_values[i, 0:32] against SetPhi[|i-j|], LDS ds_add_f32 combine.
// B fragments are double-buffered in registers to overlap global loads with
// the WMMA + sigmoid + FMA work of the previous tile.
// ---------------------------------------------------------------------------
__global__ __launch_bounds__(256) void attn_setvals(
    const unsigned int* __restrict__ qA, const unsigned int* __restrict__ kB,
    const float* __restrict__ SetPhi, const float* __restrict__ offset,
    float* __restrict__ set_values)
{
    __shared__ float sc[8][16][16];    // per-wave score tile
    __shared__ float svred[16][32];    // block-level set_values accumulator

    int b   = blockIdx.x >> 6;
    int it  = blockIdx.x & 63;
    int tid = threadIdx.x;
    int w   = tid >> 5;
    int lane = tid & 31;
    float off = offset[0];

    ((float*)svred)[tid]       = 0.0f;
    ((float*)svred)[tid + 256] = 0.0f;
    __syncthreads();

    // A fragment (q) for this i-tile: 16 dwords/lane, two K-chunks
    const unsigned int* ap = qA + (((b * 64) + it) * 32 + lane) * 16;
    v16h a0 = *(const v16h*)(ap);
    v16h a1 = *(const v16h*)(ap + 8);

    int i_loc = lane >> 1;             // accumulation sharding: 2 lanes per row
    int kh    = lane & 1;              // low/high 16 of set_dim
    int ig_acc = it * 16 + i_loc;
    float sv[16];
    #pragma unroll
    for (int kk = 0; kk < 16; ++kk) sv[kk] = 0.0f;

    const float inv_sqrt_dk = 0.125f;  // 1/sqrt(64)
    int jl  = lane & 15;
    int mhi = (lane >> 4) * 8;

    // preload first B fragment (j-tile = w)
    const unsigned int* kbase_p = kB + ((b * 64) * 32 + lane) * 16;
    v16h b0 = *(const v16h*)(kbase_p + (size_t)w * 512);
    v16h b1 = *(const v16h*)(kbase_p + (size_t)w * 512 + 8);

    for (int t = 0; t < 8; ++t) {
        int jt = w + t * 8;
        v16h nb0, nb1;
        if (t < 7) {                   // prefetch next tile's fragment
            const unsigned int* bp = kbase_p + (size_t)(jt + 8) * 512;
            nb0 = *(const v16h*)(bp);
            nb1 = *(const v16h*)(bp + 8);
        }
        v8f c = {};
        c = __builtin_amdgcn_wmma_f32_16x16x32_f16(false, a0, false, b0,
                                                   (short)0, c, false, false);
        c = __builtin_amdgcn_wmma_f32_16x16x32_f16(false, a1, false, b1,
                                                   (short)0, c, false, false);
        // C layout: VGPR r, lanes 0-15 -> M=r, lanes 16-31 -> M=r+8; N=lane&15
        #pragma unroll
        for (int r = 0; r < 8; ++r) {
            int il = r + mhi;
            int ig = it * 16 + il;
            int jg = jt * 16 + jl;
            float s = fast_sigmoid(c[r] * inv_sqrt_dk + off);
            if (ig == jg) s = 0.0f;
            sc[w][il][jl] = s;
        }
        __syncthreads();
        // set_values[i, k] += att(i,j) * SetPhi[|i-j|, k]
        #pragma unroll 1
        for (int j = 0; j < 16; ++j) {
            float a_s = sc[w][i_loc][j];
            int jg = jt * 16 + j;
            int dd = (ig_acc >= jg) ? (ig_acc - jg) : (jg - ig_acc);
            const float4* sp = (const float4*)(SetPhi + dd * 32 + kh * 16);
            #pragma unroll
            for (int q4 = 0; q4 < 4; ++q4) {
                float4 p = sp[q4];
                sv[q4 * 4 + 0] += a_s * p.x;
                sv[q4 * 4 + 1] += a_s * p.y;
                sv[q4 * 4 + 2] += a_s * p.z;
                sv[q4 * 4 + 3] += a_s * p.w;
            }
        }
        __syncthreads();
        b0 = nb0; b1 = nb1;
    }
    // combine waves (ds_add_f32)
    #pragma unroll
    for (int kk = 0; kk < 16; ++kk)
        atomicAdd(&svred[i_loc][kh * 16 + kk], sv[kk]);
    __syncthreads();
    float* dst = set_values + ((b * 1024) + it * 16) * 32;
    dst[tid]       = ((float*)svred)[tid];
    dst[tid + 256] = ((float*)svred)[tid + 256];
}

// ---------------------------------------------------------------------------
// summarized[b,j,k] = sum_i set_values[b,i,j] * v[b,i,k]   (32 x 8 per b)
// ---------------------------------------------------------------------------
__global__ __launch_bounds__(256) void summarize_kernel(
    const float* __restrict__ setv, const float* __restrict__ vbuf,
    float* __restrict__ summ)
{
    int b = blockIdx.x, tid = threadIdx.x;
    int j = tid >> 3, k = tid & 7;
    float acc = 0.0f;
    const float* sp = setv + (b * 1024) * 32 + j;
    const float* vp = vbuf + (b * 1024) * 8 + k;
    for (int i = 0; i < 1024; ++i)
        acc += sp[i * 32] * vp[i * 8];
    summ[b * 256 + tid] = acc;
}

// ---------------------------------------------------------------------------
// Head MLP 256->256->256->4 + softmax, one block per batch.
// ---------------------------------------------------------------------------
__global__ __launch_bounds__(256) void head_kernel(
    const float* __restrict__ summ,
    const float* __restrict__ HW1, const float* __restrict__ Hb1,
    const float* __restrict__ HW2, const float* __restrict__ Hb2,
    const float* __restrict__ HW3, const float* __restrict__ Hb3,
    float* __restrict__ out)
{
    __shared__ float buf[256], h[256], lg[4];
    int b = blockIdx.x, tid = threadIdx.x;
    buf[tid] = summ[b * 256 + tid];
    __syncthreads();
    float acc = Hb1[tid];
    { const float* w = HW1 + tid * 256;
      for (int d = 0; d < 256; ++d) acc += w[d] * buf[d]; }
    h[tid] = fmaxf(acc, 0.0f);
    __syncthreads();
    acc = Hb2[tid];
    { const float* w = HW2 + tid * 256;
      for (int d = 0; d < 256; ++d) acc += w[d] * h[d]; }
    buf[tid] = fmaxf(acc, 0.0f);
    __syncthreads();
    if (tid < 4) {
        float a = Hb3[tid]; const float* w = HW3 + tid * 256;
        for (int d = 0; d < 256; ++d) a += w[d] * buf[d];
        lg[tid] = a;
    }
    __syncthreads();
    if (tid == 0) {
        float m = fmaxf(fmaxf(lg[0], lg[1]), fmaxf(lg[2], lg[3]));
        float e0 = __expf(lg[0] - m), e1 = __expf(lg[1] - m);
        float e2 = __expf(lg[2] - m), e3 = __expf(lg[3] - m);
        float rs = __builtin_amdgcn_rcpf(e0 + e1 + e2 + e3);
        out[b * 4 + 0] = e0 * rs; out[b * 4 + 1] = e1 * rs;
        out[b * 4 + 2] = e2 * rs; out[b * 4 + 3] = e3 * rs;
    }
}

extern "C" void kernel_launch(void* const* d_in, const int* in_sizes, int n_in,
                              void* d_out, int out_size, void* d_ws, size_t ws_size,
                              hipStream_t stream) {
    const float* x   = (const float*)d_in[0];
    const float* WQ  = (const float*)d_in[1];
    const float* WK  = (const float*)d_in[2];
    const float* VW1 = (const float*)d_in[3];
    const float* Vb1 = (const float*)d_in[4];
    const float* VW2 = (const float*)d_in[5];
    const float* Vb2 = (const float*)d_in[6];
    const float* VW3 = (const float*)d_in[7];
    const float* Vb3 = (const float*)d_in[8];
    const float* PW1 = (const float*)d_in[9];
    const float* Pb1 = (const float*)d_in[10];
    const float* PW2 = (const float*)d_in[11];
    const float* Pb2 = (const float*)d_in[12];
    const float* PW3 = (const float*)d_in[13];
    const float* Pb3 = (const float*)d_in[14];
    const float* PW4 = (const float*)d_in[15];
    const float* Pb4 = (const float*)d_in[16];
    const float* offset = (const float*)d_in[17];
    const float* HW1 = (const float*)d_in[18];
    const float* Hb1 = (const float*)d_in[19];
    const float* HW2 = (const float*)d_in[20];
    const float* Hb2 = (const float*)d_in[21];
    const float* HW3 = (const float*)d_in[22];
    const float* Hb3 = (const float*)d_in[23];

    // workspace layout (float units)
    float* ws            = (float*)d_ws;
    float* SetPhi        = ws;                               //  32768
    float* vbuf          = ws + 32768;                       //  65536
    unsigned int* qA     = (unsigned int*)(ws + 98304);      // 262144 dwords
    unsigned int* kB     = (unsigned int*)(ws + 360448);     // 262144 dwords
    float* setv          = ws + 622592;                      // 262144
    float* summ          = ws + 884736;                      //   2048
    float* outp          = (float*)d_out;                    // [8,4]

    setphi_kernel<<<16, 64, 0, stream>>>(PW1, Pb1, PW2, Pb2, PW3, Pb3, PW4, Pb4, SetPhi);
    vproj_kernel<<<32, 256, 0, stream>>>(x, VW1, Vb1, VW2, Vb2, VW3, Vb3, vbuf);
    pack_qk<<<512, 32, 0, stream>>>(x, WQ, qA);
    pack_qk<<<512, 32, 0, stream>>>(x, WK, kB);
    attn_setvals<<<512, 256, 0, stream>>>(qA, kB, SetPhi, offset, setv);
    summarize_kernel<<<8, 256, 0, stream>>>(setv, vbuf, summ);
    head_kernel<<<8, 256, 0, stream>>>(summ, HW1, Hb1, HW2, Hb2, HW3, Hb3, outp);
}
